// RoiPosFeatExtraxtor_11768210391142
// MI455X (gfx1250) — matrix operand order repaired and checked by reading, not compiled, not run on hardware
//
#include <hip/hip_runtime.h>

// ---------------------------------------------------------------------------
// Problem constants (from the reference)
// ---------------------------------------------------------------------------
#define Bb      4
#define Nn      256
#define Cc      256
#define Hh      128
#define Ww      128
#define Dd      1024
#define POOLD   7
#define SRr     2
#define IN_DIM  (Cc * POOLD * POOLD)   // 12544
#define ROWS    (Bb * Nn)              // 1024
#define KSLICES 4
#define EPSF    1e-5f

typedef __attribute__((ext_vector_type(16))) __bf16 v16bf;
typedef __attribute__((ext_vector_type(8)))  __bf16 v8bf;
typedef __attribute__((ext_vector_type(8)))  float  v8f;

// ---------------------------------------------------------------------------
// Kernel 1: ROI-align.  One block per (b, n); thread = channel.
// ---------------------------------------------------------------------------
__global__ __launch_bounds__(256) void roi_kernel(
    const float* __restrict__ feats,   // (B, C, H, W)
    const float* __restrict__ boxes,   // (B, N, 4)
    float* __restrict__ pooled)        // (ROWS, IN_DIM)
{
    const int row = blockIdx.x;        // b*N + n
    const int b   = row >> 8;

    __shared__ int   iy0[14], iy1[14], ix0[14], ix1[14];
    __shared__ float sly[14], slx[14], svy[14], svx[14];

    if (threadIdx.x < 14) {
        const int i = threadIdx.x;
        const float* bx = boxes + (size_t)row * 4;
        const float x1 = bx[0] * 0.25f, y1 = bx[1] * 0.25f;
        const float x2 = bx[2] * 0.25f, y2 = bx[3] * 0.25f;
        const float rw = fmaxf(x2 - x1, 1.0f), rh = fmaxf(y2 - y1, 1.0f);
        const float bw = rw / (float)POOLD, bh = rh / (float)POOLD;
        const float o  = ((float)i + 0.5f) / (float)SRr;

        const float yy = y1 + bh * o;
        svy[i] = (yy > -1.0f && yy < (float)Hh) ? 1.0f : 0.0f;
        float yc = fminf(fmaxf(yy, 0.0f), (float)(Hh - 1));
        int y0 = (int)floorf(yc);
        iy0[i] = y0; iy1[i] = min(y0 + 1, Hh - 1); sly[i] = yc - (float)y0;

        const float xx = x1 + bw * o;
        svx[i] = (xx > -1.0f && xx < (float)Ww) ? 1.0f : 0.0f;
        float xc = fminf(fmaxf(xx, 0.0f), (float)(Ww - 1));
        int x0 = (int)floorf(xc);
        ix0[i] = x0; ix1[i] = min(x0 + 1, Ww - 1); slx[i] = xc - (float)x0;
    }
    __syncthreads();

    const int c = threadIdx.x;
    const float* f  = feats + ((size_t)(b * Cc + c)) * (Hh * Ww);
    float*       po = pooled + (size_t)row * IN_DIM + (size_t)c * (POOLD * POOLD);

    #pragma unroll
    for (int iy = 0; iy < POOLD; ++iy) {
        #pragma unroll
        for (int ix = 0; ix < POOLD; ++ix) {
            float s = 0.0f;
            #pragma unroll
            for (int sy = 0; sy < SRr; ++sy) {
                #pragma unroll
                for (int sx = 0; sx < SRr; ++sx) {
                    const int j = iy * SRr + sy;
                    const int i = ix * SRr + sx;
                    if (svy[j] * svx[i] != 0.0f) {
                        const float ly = sly[j], lx = slx[i];
                        const float hy = 1.0f - ly, hx = 1.0f - lx;
                        const int y0 = iy0[j], y1 = iy1[j];
                        const int x0 = ix0[i], x1 = ix1[i];
                        s += hy * hx * f[y0 * Ww + x0] + hy * lx * f[y0 * Ww + x1]
                           + ly * hx * f[y1 * Ww + x0] + ly * lx * f[y1 * Ww + x1];
                    }
                }
            }
            po[iy * POOLD + ix] = s * 0.25f;
        }
    }
}

// ---------------------------------------------------------------------------
// Split-K partial GEMM, fp32-faithful via bf16 hi/lo split on
// v_wmma_f32_16x16x32_bf16:   Cpart[slice] = A[:, ks:ks+klen] @ B[ks:ks+klen, :]
// Block: 256 threads (8 waves), 128x128 tile, K-steps of 32, double-buffered
// LDS (2 x 40KB) so tile i+1 streams in while WMMAs consume tile i.
// grid = (N/128, M/128, KSLICES) -> 256 workgroups: fills the MI455X WGP array.
// ---------------------------------------------------------------------------
__global__ __launch_bounds__(256) void gemm_split_bf16(
    const float* __restrict__ A, const float* __restrict__ B,
    float* __restrict__ Cpart,          // KSLICES consecutive MxN slices
    int M, int N, int K, int klen)
{
    constexpr int KP = 40;              // halfs per LDS row (32 + 8 pad)
    __shared__ __bf16 Ah[2][128 * KP];
    __shared__ __bf16 Al[2][128 * KP];
    __shared__ __bf16 Bh[2][128 * KP];
    __shared__ __bf16 Bl[2][128 * KP];

    const int t     = threadIdx.x;
    const int m0    = blockIdx.y * 128;
    const int n0    = blockIdx.x * 128;
    const int slice = blockIdx.z;
    const int ks    = slice * klen;
    const int wave  = t >> 5;
    const int lane  = t & 31;
    const int lrow  = lane & 15;
    const int hsel  = lane >> 4;

    float* __restrict__ Cs = Cpart + (size_t)slice * M * N;

    // Stage one 128x32 A-tile and 32x128 B-tile (N-major) as bf16 hi/lo.
    auto stage = [&](int buf, int kpos) {
        #pragma unroll
        for (int it = 0; it < 4; ++it) {
            const int idx = t + it * 256;            // 1024 float4 chunks each
            {
                const int r  = idx >> 3;
                const int cf = (idx & 7) * 4;
                const float4 va = *(const float4*)(A + (size_t)(m0 + r) * K + kpos + cf);
                const float av[4] = {va.x, va.y, va.z, va.w};
                #pragma unroll
                for (int j = 0; j < 4; ++j) {
                    const __bf16 h = (__bf16)av[j];
                    Ah[buf][r * KP + cf + j] = h;
                    Al[buf][r * KP + cf + j] = (__bf16)(av[j] - (float)h);
                }
            }
            {
                const int kk = idx >> 5;
                const int nf = (idx & 31) * 4;
                const float4 vb = *(const float4*)(B + (size_t)(kpos + kk) * N + n0 + nf);
                const float bv[4] = {vb.x, vb.y, vb.z, vb.w};
                #pragma unroll
                for (int j = 0; j < 4; ++j) {
                    const __bf16 h = (__bf16)bv[j];
                    Bh[buf][(nf + j) * KP + kk] = h;
                    Bl[buf][(nf + j) * KP + kk] = (__bf16)(bv[j] - (float)h);
                }
            }
        }
    };

    v8f acc[8];
    #pragma unroll
    for (int i = 0; i < 8; ++i)
        acc[i] = (v8f){0.f, 0.f, 0.f, 0.f, 0.f, 0.f, 0.f, 0.f};

    const int nsteps = klen >> 5;
    stage(0, ks);
    int p = 0;

    for (int i = 0; i < nsteps; ++i) {
        __syncthreads();                 // buffer p ready (s_wait_dscnt 0 + barrier)
        if (i + 1 < nsteps) stage(p ^ 1, ks + (i + 1) * 32);   // overlap with WMMAs

        // ---- A fragments (hi/lo): ISA 7.12.2 interleaved lane-half K layout ----
        const int arow = wave * 16 + lrow;
        const v8bf a0h = *(const v8bf*)&Ah[p][arow * KP + hsel * 8];
        const v8bf a1h = *(const v8bf*)&Ah[p][arow * KP + 16 + hsel * 8];
        const v8bf a0l = *(const v8bf*)&Al[p][arow * KP + hsel * 8];
        const v8bf a1l = *(const v8bf*)&Al[p][arow * KP + 16 + hsel * 8];
        const v16bf a_hi = __builtin_shufflevector(a0h, a1h, 0,1,2,3,4,5,6,7,8,9,10,11,12,13,14,15);
        const v16bf a_lo = __builtin_shufflevector(a0l, a1l, 0,1,2,3,4,5,6,7,8,9,10,11,12,13,14,15);

        // ---- 8 N-tiles x 3 split products; grouped by 4 to space RAW deps ----
        #pragma unroll
        for (int half = 0; half < 2; ++half) {
            v16bf bhf[4], blf[4];
            #pragma unroll
            for (int q = 0; q < 4; ++q) {
                const int nrow = (half * 4 + q) * 16 + lrow;
                const v8bf b0h = *(const v8bf*)&Bh[p][nrow * KP + hsel * 8];
                const v8bf b1h = *(const v8bf*)&Bh[p][nrow * KP + 16 + hsel * 8];
                const v8bf b0l = *(const v8bf*)&Bl[p][nrow * KP + hsel * 8];
                const v8bf b1l = *(const v8bf*)&Bl[p][nrow * KP + 16 + hsel * 8];
                bhf[q] = __builtin_shufflevector(b0h, b1h, 0,1,2,3,4,5,6,7,8,9,10,11,12,13,14,15);
                blf[q] = __builtin_shufflevector(b0l, b1l, 0,1,2,3,4,5,6,7,8,9,10,11,12,13,14,15);
            }
            #pragma unroll
            for (int q = 0; q < 4; ++q)
                acc[half * 4 + q] = __builtin_amdgcn_wmma_f32_16x16x32_bf16(
                    false, a_hi, false, bhf[q], (short)0, acc[half * 4 + q], false, false);
            #pragma unroll
            for (int q = 0; q < 4; ++q)
                acc[half * 4 + q] = __builtin_amdgcn_wmma_f32_16x16x32_bf16(
                    false, a_lo, false, bhf[q], (short)0, acc[half * 4 + q], false, false);
            #pragma unroll
            for (int q = 0; q < 4; ++q)
                acc[half * 4 + q] = __builtin_amdgcn_wmma_f32_16x16x32_bf16(
                    false, a_hi, false, blf[q], (short)0, acc[half * 4 + q], false, false);
        }
        p ^= 1;
    }

    // ---- store partials: C/D layout -> lanes 0-15: M=r, lanes 16-31: M=8+r ----
    const int row_base = m0 + wave * 16 + hsel * 8;
    #pragma unroll
    for (int nt = 0; nt < 8; ++nt) {
        const int col = n0 + nt * 16 + lrow;
        #pragma unroll
        for (int r = 0; r < 8; ++r)
            Cs[(size_t)(row_base + r) * N + col] = acc[nt][r];
    }
}

// ---------------------------------------------------------------------------
// Epilogue for GEMM1: h1 = relu(sum_slices(Cpart1) + b1).  float4 vectorized.
// ---------------------------------------------------------------------------
__global__ __launch_bounds__(256) void reduce_relu_kernel(
    const float* __restrict__ Cpart, const float* __restrict__ bias,
    float* __restrict__ Out, int N)
{
    const int row = blockIdx.x;
    const int cg  = threadIdx.x * 4;
    const size_t base = (size_t)row * N + cg;
    const size_t ss   = (size_t)gridDim.x * N;

    float4 s = *(const float4*)(Cpart + base);
    #pragma unroll
    for (int sl = 1; sl < KSLICES; ++sl) {
        const float4 v = *(const float4*)(Cpart + sl * ss + base);
        s.x += v.x; s.y += v.y; s.z += v.z; s.w += v.w;
    }
    const float4 bz = *(const float4*)(bias + cg);
    float4 o;
    o.x = fmaxf(s.x + bz.x, 0.0f);
    o.y = fmaxf(s.y + bz.y, 0.0f);
    o.z = fmaxf(s.z + bz.z, 0.0f);
    o.w = fmaxf(s.w + bz.w, 0.0f);
    *(float4*)(Out + base) = o;
}

// ---------------------------------------------------------------------------
// Kernel 4: folds GEMM2's split-K reduction + b2, then
// bbox projection (K=4) + LN + residual add + LN, fused.  One block per row.
// ---------------------------------------------------------------------------
__global__ __launch_bounds__(256) void tail_kernel(
    const float* __restrict__ boxes, const float* __restrict__ hpart,
    const float* __restrict__ b2,
    const float* __restrict__ bbox_W, const float* __restrict__ bbox_b,
    const float* __restrict__ g1, const float* __restrict__ be1,
    const float* __restrict__ g2, const float* __restrict__ be2,
    float* __restrict__ out)
{
    const int row = blockIdx.x;
    const int t   = threadIdx.x;

    __shared__ float red[256];
    __shared__ float nb[4];

    if (t < 4) nb[t] = boxes[(size_t)row * 4 + t] * (0.25f / 128.0f);
    __syncthreads();

    auto block_sum = [&](float v) -> float {
        red[t] = v;
        __syncthreads();
        #pragma unroll
        for (int off = 128; off > 0; off >>= 1) {
            if (t < off) red[t] += red[t + off];
            __syncthreads();
        }
        const float r = red[0];
        __syncthreads();
        return r;
    };

    float p[4];
    #pragma unroll
    for (int q = 0; q < 4; ++q) {
        const int j = t + q * 256;
        float s = bbox_b[j];
        #pragma unroll
        for (int k = 0; k < 4; ++k) s += nb[k] * bbox_W[k * Dd + j];
        p[q] = s;
    }

    // LayerNorm 1 (bbox branch)
    float mean = block_sum(p[0] + p[1] + p[2] + p[3]) * (1.0f / Dd);
    float vs = 0.0f;
    #pragma unroll
    for (int q = 0; q < 4; ++q) { const float d = p[q] - mean; vs += d * d; }
    float inv = 1.0f / sqrtf(block_sum(vs) * (1.0f / Dd) + EPSF);

    // h = sum of GEMM2 K-slices + b2; then residual add
    float tv[4];
    #pragma unroll
    for (int q = 0; q < 4; ++q) {
        const int j = t + q * 256;
        float hv = b2[j];
        #pragma unroll
        for (int sl = 0; sl < KSLICES; ++sl)
            hv += hpart[(size_t)sl * ROWS * Dd + (size_t)row * Dd + j];
        const float pn = (p[q] - mean) * inv * g1[j] + be1[j];
        tv[q] = hv + pn;
    }

    // LayerNorm 2 (residual)
    float mean2 = block_sum(tv[0] + tv[1] + tv[2] + tv[3]) * (1.0f / Dd);
    float vs2 = 0.0f;
    #pragma unroll
    for (int q = 0; q < 4; ++q) { const float d = tv[q] - mean2; vs2 += d * d; }
    float inv2 = 1.0f / sqrtf(block_sum(vs2) * (1.0f / Dd) + EPSF);

    #pragma unroll
    for (int q = 0; q < 4; ++q) {
        const int j = t + q * 256;
        out[(size_t)row * Dd + j] = (tv[q] - mean2) * inv2 * g2[j] + be2[j];
    }
}

// ---------------------------------------------------------------------------
// Launcher. ws layout (bytes):
//   pooled  : ROWS x IN_DIM f32                       = 51.4 MB
//   Cpart1  : KSLICES x ROWS x D f32                  = 16  MB
//   h1      : ROWS x D f32                            =  4  MB
//   Cpart2  : aliases pooled (dead after GEMM1)       = 16  MB (in-place)
// Every byte consumed is re-written each call (poison-safe, deterministic).
// ---------------------------------------------------------------------------
extern "C" void kernel_launch(void* const* d_in, const int* in_sizes, int n_in,
                              void* d_out, int out_size, void* d_ws, size_t ws_size,
                              hipStream_t stream)
{
    const float* feats  = (const float*)d_in[0];
    const float* boxes  = (const float*)d_in[1];
    // d_in[2] = img_sizes (int32) — unused by the math path
    const float* W1     = (const float*)d_in[3];
    const float* b1     = (const float*)d_in[4];
    const float* W2     = (const float*)d_in[5];
    const float* b2     = (const float*)d_in[6];
    const float* bboxW  = (const float*)d_in[7];
    const float* bboxb  = (const float*)d_in[8];
    const float* lnbg   = (const float*)d_in[9];
    const float* lnbb   = (const float*)d_in[10];
    const float* lnag   = (const float*)d_in[11];
    const float* lnab   = (const float*)d_in[12];

    float* pooled = (float*)d_ws;
    float* Cpart1 = pooled + (size_t)ROWS * IN_DIM;
    float* h1     = Cpart1 + (size_t)KSLICES * ROWS * Dd;
    float* Cpart2 = pooled;                         // alias: pooled dead after GEMM1

    roi_kernel<<<ROWS, 256, 0, stream>>>(feats, boxes, pooled);

    // GEMM1 partials: [1024 x 12544 x 1024], 4 K-slices of 3136 -> 256 WGs
    gemm_split_bf16<<<dim3(Dd / 128, ROWS / 128, KSLICES), 256, 0, stream>>>(
        pooled, W1, Cpart1, ROWS, Dd, IN_DIM, IN_DIM / KSLICES);

    // h1 = relu(sum(Cpart1) + b1)
    reduce_relu_kernel<<<ROWS, 256, 0, stream>>>(Cpart1, b1, h1, Dd);

    // GEMM2 partials: [1024 x 1024 x 1024], 4 K-slices of 256 -> 256 WGs
    gemm_split_bf16<<<dim3(Dd / 128, ROWS / 128, KSLICES), 256, 0, stream>>>(
        h1, W2, Cpart2, ROWS, Dd, Dd, Dd / KSLICES);

    // fused: sum(Cpart2)+b2, bbox proj, LN, residual, LN
    tail_kernel<<<ROWS, 256, 0, stream>>>(boxes, Cpart2, b2, bboxW, bboxb,
                                          lnbg, lnbb, lnag, lnab,
                                          (float*)d_out);
}
